// QRNNLayer_23845658427812
// MI455X (gfx1250) — compile-verified, matrix operand order checked by reading, not run on hardware
//
#include <hip/hip_runtime.h>
#include <hip/hip_bf16.h>
#include <math.h>

typedef __attribute__((ext_vector_type(16))) __bf16 bf16x16;
typedef __attribute__((ext_vector_type(8)))  __bf16 bf16x8;
typedef __attribute__((ext_vector_type(4)))  __bf16 bf16x4;
typedef __attribute__((ext_vector_type(2)))  __bf16 bf16x2;
typedef __attribute__((ext_vector_type(8)))  float  f32x8;
typedef __attribute__((ext_vector_type(4)))  float  f32x4;

#define QB   8
#define QS   4096
#define QD   512
#define QH   512
#define QN3  (3 * QH)          // 1536
#define QM   (QB * QS)         // 32768
#define NSEG 32
#define SEGL (QS / NSEG)       // 128
#define NKSTEPS (QD / 32)      // 16

__device__ __forceinline__ void sched_fence() {
#if __has_builtin(__builtin_amdgcn_sched_barrier)
    __builtin_amdgcn_sched_barrier(0);
#endif
}

// ---------------------------------------------------------------------------
// CDNA5 hardware tanh (TRANS pipe). sigmoid via exact identity.
// ---------------------------------------------------------------------------
__device__ __forceinline__ float fast_tanh(float x) {
#if __has_builtin(__builtin_amdgcn_tanhf)
    return __builtin_amdgcn_tanhf(x);
#elif __has_builtin(__builtin_amdgcn_tanh_f32)
    return __builtin_amdgcn_tanh_f32(x);
#else
    float y;
    asm volatile("v_tanh_f32 %0, %1" : "=v"(y) : "v"(x));
    return y;
#endif
}

__device__ __forceinline__ float fast_sigmoid(float x) {
    return 0.5f * fast_tanh(0.5f * x) + 0.5f;
}

// ---------------------------------------------------------------------------
// Fragment loader: 16x32 bf16 tile, row-major [16][32] in LDS.
// Lane l -> row = l%16, K-half base = (l/16)*8; VGPR0..3 = K base+0..7,
// VGPR4..7 = K 16+base+0..7  => two 16-byte LDS loads per lane.
// ---------------------------------------------------------------------------
__device__ __forceinline__ bf16x16 load_frag16x32(const __bf16* base) {
    int lane = threadIdx.x & 31;
    int row  = lane & 15;
    int koff = (lane >> 4) * 8;
    const __bf16* p = base + row * 32 + koff;
    bf16x8 lo = *(const bf16x8*)(p);
    bf16x8 hi = *(const bf16x8*)(p + 16);
    return __builtin_shufflevector(lo, hi, 0, 1, 2, 3, 4, 5, 6, 7,
                                           8, 9, 10, 11, 12, 13, 14, 15);
}

// ---------------------------------------------------------------------------
// Kernel 1: y = X @ W^T + b, fused gate activations, double-buffered LDS
// with register prefetch pinned above the WMMA block.
//   gate 0:  z = tanh(y)    -> zf[2*idx+0] (bf16)
//   gate 1:  f = sigmoid(y) -> zf[2*idx+1] (bf16)
//   gate 2:  o = sigmoid(y) -> obf[idx]    (bf16)
// ---------------------------------------------------------------------------
__global__ void qrnn_gemm_gates(const float* __restrict__ X,
                                const float* __restrict__ W,
                                const float* __restrict__ bias,
                                __bf16* __restrict__ zf,
                                __bf16* __restrict__ obf) {
    __shared__ __align__(16) __bf16 As[2][128 * 32];
    __shared__ __align__(16) __bf16 Bs[2][64 * 32];

    const int t     = threadIdx.x;          // 0..255
    const int mBase = blockIdx.y * 128;
    const int nBase = blockIdx.x * 64;
    const int wave  = t >> 5;               // 0..7
    const int wm    = wave & 3;             // 4 M-groups of 32
    const int wn    = wave >> 2;            // 2 N-groups of 32

    f32x8 acc[2][2];
    #pragma unroll
    for (int i = 0; i < 2; ++i)
        #pragma unroll
        for (int j = 0; j < 2; ++j)
            acc[i][j] = (f32x8)0.0f;

    f32x4 ra[4], rb[2];

    auto global_fetch = [&](int k0) {
        #pragma unroll
        for (int p = 0; p < 4; ++p) {
            int idx = t + p * 256;
            int row = idx >> 3;
            int c4  = idx & 7;
            ra[p] = *(const f32x4*)(X + (size_t)(mBase + row) * QD + k0 + c4 * 4);
        }
        #pragma unroll
        for (int p = 0; p < 2; ++p) {
            int idx = t + p * 256;
            int row = idx >> 3;
            int c4  = idx & 7;
            rb[p] = *(const f32x4*)(W + (size_t)(nBase + row) * QD + k0 + c4 * 4);
        }
    };

    auto lds_store = [&](int buf) {
        #pragma unroll
        for (int p = 0; p < 4; ++p) {
            int idx = t + p * 256;
            int row = idx >> 3;
            int c4  = idx & 7;
            bf16x4 bv = { (__bf16)ra[p].x, (__bf16)ra[p].y,
                          (__bf16)ra[p].z, (__bf16)ra[p].w };
            *(bf16x4*)(&As[buf][row * 32 + c4 * 4]) = bv;
        }
        #pragma unroll
        for (int p = 0; p < 2; ++p) {
            int idx = t + p * 256;
            int row = idx >> 3;
            int c4  = idx & 7;
            bf16x4 bv = { (__bf16)rb[p].x, (__bf16)rb[p].y,
                          (__bf16)rb[p].z, (__bf16)rb[p].w };
            *(bf16x4*)(&Bs[buf][row * 32 + c4 * 4]) = bv;
        }
    };

    global_fetch(0);
    lds_store(0);
    __syncthreads();

    for (int k = 0; k < NKSTEPS; ++k) {
        int cur = k & 1;

        // Prefetch next tile's global data; pin the loads ABOVE the WMMA
        // block so they stay in flight underneath the matrix work.
        if (k + 1 < NKSTEPS) global_fetch((k + 1) * 32);
        sched_fence();

        bf16x16 a0 = load_frag16x32(&As[cur][(wm * 32 + 0)  * 32]);
        bf16x16 a1 = load_frag16x32(&As[cur][(wm * 32 + 16) * 32]);
        bf16x16 b0 = load_frag16x32(&Bs[cur][(wn * 32 + 0)  * 32]);
        bf16x16 b1 = load_frag16x32(&Bs[cur][(wn * 32 + 16) * 32]);

        acc[0][0] = __builtin_amdgcn_wmma_f32_16x16x32_bf16(false, a0, false, b0,
                        (short)0, acc[0][0], false, false);
        acc[0][1] = __builtin_amdgcn_wmma_f32_16x16x32_bf16(false, a0, false, b1,
                        (short)0, acc[0][1], false, false);
        acc[1][0] = __builtin_amdgcn_wmma_f32_16x16x32_bf16(false, a1, false, b0,
                        (short)0, acc[1][0], false, false);
        acc[1][1] = __builtin_amdgcn_wmma_f32_16x16x32_bf16(false, a1, false, b1,
                        (short)0, acc[1][1], false, false);

        // Keep the converts/stores (first USES of the prefetched data)
        // below the WMMAs: the load-wait lands here, after the compute.
        sched_fence();

        if (k + 1 < NKSTEPS) {
            lds_store(cur ^ 1);
            __syncthreads();
        }
    }

    // Epilogue. C/D layout: VGPR r, lanes 0-15 -> (M=r, N=lane);
    // lanes 16-31 -> (M=r+8, N=lane-16).
    const int lane    = t & 31;
    const int n       = lane & 15;
    const int rowhalf = lane >> 4;

    #pragma unroll
    for (int i = 0; i < 2; ++i) {
        #pragma unroll
        for (int j = 0; j < 2; ++j) {
            int m0   = mBase + wm * 32 + i * 16;
            int n0   = nBase + wn * 32 + j * 16;   // uniform per tile
            int gate = n0 >> 9;                    // 0=z, 1=f, 2=o
            int ncol = n0 + n;
            int nh   = ncol - (gate << 9);
            float bv = bias[ncol];
            #pragma unroll
            for (int r = 0; r < 8; ++r) {
                int   m   = m0 + r + rowhalf * 8;
                float val = acc[i][j][r] + bv;
                int   idx = m * QH + nh;
                if (gate == 0)      zf[2 * idx + 0] = (__bf16)fast_tanh(val);
                else if (gate == 1) zf[2 * idx + 1] = (__bf16)fast_sigmoid(val);
                else                obf[idx]        = (__bf16)fast_sigmoid(val);
            }
        }
    }
}

// ---------------------------------------------------------------------------
// Scan pass 1: per (b, h, seg) compute segment composition of
// h <- f*z + (1-f)*h  ==  h_end = A*h_start + C,  A = prod(1-f).
// ---------------------------------------------------------------------------
__global__ void qrnn_scan_seg(const __bf16* __restrict__ zf,
                              float* __restrict__ segA,
                              float* __restrict__ segC) {
    int tid  = blockIdx.x * blockDim.x + threadIdx.x;   // 0..B*NSEG*H-1
    int h    = tid & (QH - 1);
    int rest = tid >> 9;
    int seg  = rest & (NSEG - 1);
    int b    = rest >> 5;
    size_t base = ((size_t)(b * QS + seg * SEGL)) * QH + h;

    float A = 1.0f, C = 0.0f;
    #pragma unroll 4
    for (int tt = 0; tt < SEGL; ++tt) {
        bf16x2 p = *(const bf16x2*)(zf + 2 * (base + (size_t)tt * QH));
        float z   = (float)p.x;
        float f   = (float)p.y;
        float omf = 1.0f - f;
        C = f * z + omf * C;
        A *= omf;
    }
    int sidx = (b * NSEG + seg) * QH + h;
    segA[sidx] = A;
    segC[sidx] = C;
}

// ---------------------------------------------------------------------------
// Scan pass 2: per (b, h) serially combine 32 segment summaries; record each
// segment's incoming state; write final state c[:, -1].
// ---------------------------------------------------------------------------
__global__ void qrnn_scan_combine(const float* __restrict__ segA,
                                  const float* __restrict__ segC,
                                  float* __restrict__ segH,
                                  float* __restrict__ out) {
    int tid = blockIdx.x * blockDim.x + threadIdx.x;    // 0..B*H-1
    int h   = tid & (QH - 1);
    int b   = tid >> 9;
    float hprev = 0.0f;
    #pragma unroll
    for (int seg = 0; seg < NSEG; ++seg) {
        int idx = (b * NSEG + seg) * QH + h;
        segH[idx] = hprev;
        hprev = segA[idx] * hprev + segC[idx];
    }
    out[(size_t)QB * QS * QH + b * QH + h] = hprev;     // c_gate[:, -1]
}

// ---------------------------------------------------------------------------
// Scan pass 3: replay each segment from its true start state; fuse output
// gate: out = sigmoid(o) * c.
// ---------------------------------------------------------------------------
__global__ void qrnn_scan_apply(const __bf16* __restrict__ zf,
                                const __bf16* __restrict__ obf,
                                const float* __restrict__ segH,
                                float* __restrict__ out) {
    int tid  = blockIdx.x * blockDim.x + threadIdx.x;
    int h    = tid & (QH - 1);
    int rest = tid >> 9;
    int seg  = rest & (NSEG - 1);
    int b    = rest >> 5;
    size_t base = ((size_t)(b * QS + seg * SEGL)) * QH + h;

    float hc = segH[(b * NSEG + seg) * QH + h];
    #pragma unroll 4
    for (int tt = 0; tt < SEGL; ++tt) {
        size_t idx = base + (size_t)tt * QH;
        bf16x2 p = *(const bf16x2*)(zf + 2 * idx);
        float z = (float)p.x;
        float f = (float)p.y;
        hc = f * z + (1.0f - f) * hc;
        out[idx] = (float)obf[idx] * hc;
    }
}

// ---------------------------------------------------------------------------
extern "C" void kernel_launch(void* const* d_in, const int* in_sizes, int n_in,
                              void* d_out, int out_size, void* d_ws, size_t ws_size,
                              hipStream_t stream) {
    const float* X    = (const float*)d_in[0];   // [B, S, D]
    const float* W    = (const float*)d_in[1];   // [3H, D]
    const float* bias = (const float*)d_in[2];   // [3H]
    float* out = (float*)d_out;                  // [B,S,H] ++ [B,H]

    // Workspace carve-out:
    //   zf:  interleaved (z,f) pairs, 2*M*H bf16 = 64 MB
    //   obf: sigmoid(o), M*H bf16 = 32 MB
    //   segA/segC/segH: B*NSEG*H fp32 = 0.5 MB each
    char* ws = (char*)d_ws;
    __bf16* zf  = (__bf16*)ws;
    __bf16* obf = (__bf16*)(ws + 2 * (size_t)QM * QH * sizeof(__bf16));
    float*  segA = (float*)(ws + 3 * (size_t)QM * QH * sizeof(__bf16));
    float*  segC = segA + (size_t)QB * NSEG * QH;
    float*  segH = segC + (size_t)QB * NSEG * QH;

    dim3 gemmGrid(QN3 / 64, QM / 128);           // (24, 256)
    qrnn_gemm_gates<<<gemmGrid, 256, 0, stream>>>(X, W, bias, zf, obf);

    int segThreads = QB * NSEG * QH;             // 131072
    qrnn_scan_seg<<<segThreads / 256, 256, 0, stream>>>(zf, segA, segC);

    int chThreads = QB * QH;                     // 4096
    qrnn_scan_combine<<<chThreads / 256, 256, 0, stream>>>(segA, segC, segH, out);

    qrnn_scan_apply<<<segThreads / 256, 256, 0, stream>>>(zf, obf, segH, out);
}
